// SimST_64699387347654
// MI455X (gfx1250) — compile-verified, matrix operand order checked by reading, not run on hardware
//
#include <hip/hip_runtime.h>
#include <hip/hip_bf16.h>

// ---------------------------------------------------------------------------
// Types
// ---------------------------------------------------------------------------
typedef __bf16 bf16_t;
typedef __attribute__((ext_vector_type(16))) __bf16 v16bf;
typedef __attribute__((ext_vector_type(8)))  float  v8f;

__device__ __forceinline__ bf16_t f2bf(float f) {
    unsigned int u = __builtin_bit_cast(unsigned int, f);
    unsigned int r = (u + 0x7FFFu + ((u >> 16) & 1u)) >> 16;
    unsigned short s = (unsigned short)r;
    return __builtin_bit_cast(bf16_t, s);
}
__device__ __forceinline__ float bf2f(bf16_t b) {
    unsigned short s = __builtin_bit_cast(unsigned short, b);
    unsigned int u = ((unsigned int)s) << 16;
    return __builtin_bit_cast(float, u);
}

// Model constants
#define BATCH 512
#define HIST  72
#define PRED  24
#define TT    96      // HIST + PRED
#define SS    6
#define INDIM 27
#define HH    256
#define KK    5
#define DE    64
#define TK    4
#define RF    7       // 2*(TK-1)+1
#define WROWS 31      // RF + PRED ring buffer rows

// fragment-major k offset within a 32-K tile for (lane, j)
__host__ __device__ __forceinline__ int frag_koff(int lane, int j) {
    int khalf = (lane >> 4) * 8;
    return (j < 8) ? (khalf + j) : (16 + khalf + (j - 8));
}

// ---------------------------------------------------------------------------
// A-matrix gather (fused im2col) for the WMMA GEMM
// ---------------------------------------------------------------------------
struct GP {
    const bf16_t* A;   // bf16 source
    const float*  Af;  // f32 source (nca0 conv1 reads raw features)
    int lda;
    int step;          // decoder step
};

template<int MODE>
__device__ __forceinline__ bf16_t gatherA(const GP& g, int m, int k) {
    if constexpr (MODE == 0) {                 // dense bf16 row-major
        return g.A[(size_t)m * g.lda + k];
    } else if constexpr (MODE == 1) {          // nca0 conv1: feats (B,S,T,27) f32, dil=1, all s
        int s  = m % SS;  int bt = m / SS;
        int t  = bt % TT; int b  = bt / TT;
        int tap = k >> 5, ch = k & 31;
        int sp = s - 2 + tap;
        if (ch < INDIM && sp >= 0 && sp < SS)
            return f2bf(g.Af[(((size_t)b * SS + sp) * TT + t) * INDIM + ch]);
        return f2bf(0.f);
    } else if constexpr (MODE == 2) {          // nca0 conv2: in (bt,6,256), out s in {1,3,5}, dil=1
        int si = m % 3; int bt = m / 3;
        int tap = k >> 8, ch = k & 255;
        int sp = (2 * si + 1) - 2 + tap;
        if (sp >= 0 && sp < SS)
            return g.A[(((size_t)bt * SS + sp) << 8) + ch];
        return f2bf(0.f);
    } else if constexpr (MODE == 3) {          // nca1 conv1: in (bt,3,256) at s {1,3,5}, dil=2
        int si = m % 3; int bt = m / 3;
        int tap = k >> 8, ch = k & 255;
        int sp = (2 * si + 1) - 4 + 2 * tap;   // odd
        if (sp >= 1 && sp <= 5)
            return g.A[(((size_t)bt * 3 + ((sp - 1) >> 1)) << 8) + ch];
        return f2bf(0.f);
    } else if constexpr (MODE == 4) {          // nca1 conv2: out s=5 only; taps 0..2 -> si
        int tap = k >> 8, ch = k & 255;
        if (tap < 3)
            return g.A[(((size_t)m * 3 + tap) << 8) + ch];
        return f2bf(0.f);
    } else if constexpr (MODE == 5) {          // decoder conv1: window ring (B,31,512)
        int pos = m & 3; int b = m >> 2;
        int tap = k >> 9, ch = k & 511;
        return g.A[(((size_t)b * WROWS + g.step + pos + tap) << 9) + ch];
    } else {                                   // MODE 6: decoder conv2: h1 (B,4,256)
        int tap = k >> 8, ch = k & 255;
        return g.A[(((size_t)m * TK + tap) << 8) + ch];
    }
}

// ---------------------------------------------------------------------------
// WMMA GEMM: C = act(gatherA(M x Kdim) * W(Kdim x N) + bias), out bf16
//  - Block tile 128x64x32, 256 threads = 8 waves; each wave owns a 32x32
//    output (2 A-frags x 2 B-frags = 4 v_wmma_f32_16x16x32_bf16 per K-step).
//  - A staged into double-buffered LDS (single barrier per K-step, staging of
//    tile k+1 overlaps WMMA on tile k).
//  - W is pre-packed fragment-major (per-lane v16bf in the ISA B layout), so
//    B fragments are direct L2-resident global_load_b128 pairs, no LDS.
// Requires M%128==0, N%64==0, Kdim%32==0 (true for every stage here).
// ---------------------------------------------------------------------------
template<int MODE, int ACT>
__global__ __launch_bounds__(256)
void gemm_kernel(GP g, const bf16_t* __restrict__ W, const float* __restrict__ bias,
                 bf16_t* __restrict__ out, int Kdim, int N)
{
    __shared__ bf16_t As[2][128 * 32];   // [m][k], double buffered

    const int tid  = threadIdx.x;
    const int lane = tid & 31;
    const int wave = tid >> 5;
    const int wm   = wave >> 1;          // 0..3 (32-row strip of 128)
    const int wn   = wave & 1;           // 0..1 (32-col strip of 64)
    const int m0   = blockIdx.x * 128;
    const int n0   = blockIdx.y * 64;
    const int half = lane >> 4;          // K-half select per WMMA layout
    const int mrow = lane & 15;
    const int NT   = N >> 4;             // 16-wide column tiles in W
    const int nt0  = (n0 >> 4) + wn * 2; // this wave's first B column tile

    const v16bf* Wf = (const v16bf*)W;   // fragment-major: [(kt*NT+nt)*32 + lane]

    v8f acc00 = {0.f,0.f,0.f,0.f,0.f,0.f,0.f,0.f};
    v8f acc01 = acc00, acc10 = acc00, acc11 = acc00;

    auto stage = [&](int ktile, bf16_t* dst) {
        int row = tid >> 1;
        int cb  = (tid & 1) * 16;
        #pragma unroll
        for (int i = 0; i < 16; ++i)
            dst[row * 32 + cb + i] = gatherA<MODE>(g, m0 + row, ktile * 32 + cb + i);
    };

    const int nk = Kdim >> 5;
    stage(0, As[0]);
    __syncthreads();

    int p = 0;
    for (int kt = 0; kt < nk; ++kt) {
        if (kt + 1 < nk) {
            __builtin_prefetch(&Wf[((size_t)(kt + 1) * NT + nt0) * 32], 0, 3);
            stage(kt + 1, As[p ^ 1]);
        }
        // B fragments straight from L2 (fragment-major packed)
        v16bf b0 = Wf[((size_t)kt * NT + nt0 + 0) * 32 + lane];
        v16bf b1 = Wf[((size_t)kt * NT + nt0 + 1) * 32 + lane];
        // A fragments from LDS (contiguous 16B halves -> ds_load_b128)
        union { v16bf v; unsigned int u[8]; } a0, a1;
        #pragma unroll
        for (int q = 0; q < 8; ++q) {
            int kk = ((q < 4) ? 2 * q : 16 + 2 * (q - 4)) + half * 8;
            a0.u[q] = *(const unsigned int*)&As[p][(wm * 32 +      mrow) * 32 + kk];
            a1.u[q] = *(const unsigned int*)&As[p][(wm * 32 + 16 + mrow) * 32 + kk];
        }
        acc00 = __builtin_amdgcn_wmma_f32_16x16x32_bf16(false, a0.v, false, b0, (short)0, acc00, false, false);
        acc01 = __builtin_amdgcn_wmma_f32_16x16x32_bf16(false, a0.v, false, b1, (short)0, acc01, false, false);
        acc10 = __builtin_amdgcn_wmma_f32_16x16x32_bf16(false, a1.v, false, b0, (short)0, acc10, false, false);
        acc11 = __builtin_amdgcn_wmma_f32_16x16x32_bf16(false, a1.v, false, b1, (short)0, acc11, false, false);
        __syncthreads();
        p ^= 1;
    }

    // Epilogue: C layout VGPR r -> M = base + r + 8*half, N = base + (lane&15)
    int colA = n0 + wn * 32 + mrow;
    int colB = colA + 16;
    float ba = bias ? bias[colA] : 0.f;
    float bb = bias ? bias[colB] : 0.f;
    #pragma unroll
    for (int r = 0; r < 8; ++r) {
        int rowA = m0 + wm * 32 + r + half * 8;
        int rowB = rowA + 16;
        float v00 = acc00[r] + ba, v01 = acc01[r] + bb;
        float v10 = acc10[r] + ba, v11 = acc11[r] + bb;
        if constexpr (ACT == 1) {
            v00 = tanhf(v00); v01 = tanhf(v01); v10 = tanhf(v10); v11 = tanhf(v11);
        } else if constexpr (ACT == 2) {
            v00 = fmaxf(v00, 0.f); v01 = fmaxf(v01, 0.f);
            v10 = fmaxf(v10, 0.f); v11 = fmaxf(v11, 0.f);
        }
        out[(size_t)rowA * N + colA] = f2bf(v00);
        out[(size_t)rowA * N + colB] = f2bf(v01);
        out[(size_t)rowB * N + colA] = f2bf(v10);
        out[(size_t)rowB * N + colB] = f2bf(v11);
    }
}

// ---------------------------------------------------------------------------
// Weight packing into fragment-major B layout:
//   dst[((kt*NT + nt)*32 + lane)*16 + j] = W[kt*32 + frag_koff(lane,j)][nt*16 + (lane&15)]
// ---------------------------------------------------------------------------
__global__ void pack_dense_frag_kernel(bf16_t* dst, const float* src, int Kdim, int N) {
    int idx = blockIdx.x * 256 + threadIdx.x;
    if (idx >= Kdim * N) return;
    int j    = idx & 15;
    int lane = (idx >> 4) & 31;
    int rem  = idx >> 9;
    int NT   = N >> 4;
    int nt   = rem % NT;
    int kt   = rem / NT;
    int k = kt * 32 + frag_koff(lane, j);
    int n = nt * 16 + (lane & 15);
    dst[idx] = f2bf(src[(size_t)k * N + n]);
}

// conv weight (O=N, I, TKc[,1]) f32 -> fragment-major over Kdim = TKc*Cpad (zero pad i>=I)
__global__ void pack_conv_frag_kernel(bf16_t* dst, const float* src, int I, int TKc, int Cpad, int N) {
    int idx = blockIdx.x * 256 + threadIdx.x;
    int Kdim = TKc * Cpad;
    if (idx >= Kdim * N) return;
    int j    = idx & 15;
    int lane = (idx >> 4) & 31;
    int rem  = idx >> 9;
    int NT   = N >> 4;
    int nt   = rem % NT;
    int kt   = rem / NT;
    int k = kt * 32 + frag_koff(lane, j);
    int o = nt * 16 + (lane & 15);
    int tap = k / Cpad;
    int i   = k - tap * Cpad;
    float v = (i < I) ? src[((size_t)o * I + i) * TKc + tap] : 0.f;
    dst[idx] = f2bf(v);
}

// ---------------------------------------------------------------------------
// Elementwise stages
// ---------------------------------------------------------------------------
// dyn_in (B*T,256) = [tanh(loc), emb_month, emb_weekday, emb_hour]
__global__ void embed_kernel(bf16_t* dynin, const float* cl, const int* date,
                             const float* em, const float* ew, const float* eh,
                             const float* lw, const float* lb) {
    int idx = blockIdx.x * 256 + threadIdx.x;
    if (idx >= BATCH * TT * 256) return;
    int col = idx & 255;
    int bt  = idx >> 8;
    int b   = bt / TT;
    int seg = col >> 6, j = col & 63;
    float v;
    if (seg == 0)      v = tanhf(cl[b * 2] * lw[j] + cl[b * 2 + 1] * lw[64 + j] + lb[j]);
    else if (seg == 1) { int mi = date[bt * 3 + 2] - 1; v = em[mi * 64 + j]; }
    else if (seg == 2) { int wi = date[bt * 3 + 1] - 1; v = ew[wi * 64 + j]; }
    else               { int hi = date[bt * 3 + 0];     v = eh[hi * 64 + j]; }
    dynin[idx] = f2bf(v);
}

// xn (B*HIST,256) = tanh(pm25_hist @ pm25_w + pm25_b), K=2 -> VALU
__global__ void xn_kernel(bf16_t* xnb, const float* ph, const float* w, const float* bias) {
    int idx = blockIdx.x * 256 + threadIdx.x;
    if (idx >= BATCH * HIST * 256) return;
    int h = idx & 255;
    int r = idx >> 8;
    float v = tanhf(ph[(size_t)r * 2] * w[h] + ph[(size_t)r * 2 + 1] * w[256 + h] + bias[h]);
    xnb[idx] = f2bf(v);
}

// y1 (B*T*3,256) += dyn (B*T,256) broadcast over s
__global__ void add_dyn_kernel(bf16_t* y1, const bf16_t* dyn) {
    int idx = blockIdx.x * 256 + threadIdx.x;
    if (idx >= BATCH * TT * 3 * 256) return;
    int c  = idx & 255;
    int bt = (idx >> 8) / 3;
    y1[idx] = f2bf(bf2f(y1[idx]) + bf2f(dyn[(size_t)bt * 256 + c]));
}

// window ring rows 0..6 = concat(static[65..71], xn[65..71])
__global__ void init_window_kernel(bf16_t* win, const bf16_t* staticb, const bf16_t* xnb) {
    int idx = blockIdx.x * 256 + threadIdx.x;
    if (idx >= BATCH * RF * 512) return;
    int c = idx & 511;
    int r = (idx >> 9) % RF;
    int b = idx / (RF * 512);
    bf16_t v = (c < 256) ? staticb[((size_t)b * TT + (HIST - RF) + r) * 256 + c]
                         : xnb[((size_t)b * HIST + (HIST - RF) + r) * 256 + (c - 256)];
    win[((size_t)b * WROWS + r) * 512 + c] = v;
}

// per-step head: p/c dots, output store, re-embed, append window row
__global__ __launch_bounds__(256)
void finish_step_kernel(const bf16_t* pe, const float* pred_w, const float* pred_b,
                        const float* cls_w, const float* cls_b,
                        const float* pm25_w, const float* pm25_b,
                        const bf16_t* staticb, bf16_t* win, float* out, int step)
{
    __shared__ float sp[256], sc[256];
    int b = blockIdx.x, tid = threadIdx.x;
    float v = bf2f(pe[(size_t)b * 256 + tid]);
    sp[tid] = v * pred_w[tid];
    sc[tid] = v * cls_w[tid];
    __syncthreads();
    for (int s = 128; s > 0; s >>= 1) {
        if (tid < s) { sp[tid] += sp[tid + s]; sc[tid] += sc[tid + s]; }
        __syncthreads();
    }
    float p = sp[0] + pred_b[0];
    float c = sc[0] + cls_b[0];
    if (tid == 0) {
        out[((size_t)b * PRED + step) * 2 + 0] = p;
        out[((size_t)b * PRED + step) * 2 + 1] = c;
    }
    float ne = tanhf(p * pm25_w[tid] + c * pm25_w[256 + tid] + pm25_b[tid]);
    size_t wrow = ((size_t)b * WROWS + RF + step) * 512;
    win[wrow + tid]       = staticb[((size_t)b * TT + HIST + step) * 256 + tid];
    win[wrow + 256 + tid] = f2bf(ne);
}

// ---------------------------------------------------------------------------
// Host orchestration
// ---------------------------------------------------------------------------
static inline size_t alup(size_t b) { return (b + 255) & ~(size_t)255; }

extern "C" void kernel_launch(void* const* d_in, const int* in_sizes, int n_in,
                              void* d_out, int out_size, void* d_ws, size_t ws_size,
                              hipStream_t stream) {
    (void)in_sizes; (void)n_in; (void)out_size; (void)ws_size;
    const float* pm25_hist = (const float*)d_in[0];
    const float* features  = (const float*)d_in[1];
    const float* city_locs = (const float*)d_in[2];
    const int*   date_emb  = (const int*)  d_in[3];
    const float* emb_month = (const float*)d_in[4];
    const float* emb_wd    = (const float*)d_in[5];
    const float* emb_hour  = (const float*)d_in[6];
    const float* loc_w     = (const float*)d_in[7];
    const float* loc_b     = (const float*)d_in[8];
    const float* pm25_w    = (const float*)d_in[9];
    const float* pm25_b    = (const float*)d_in[10];
    const float* dyn_w1    = (const float*)d_in[11];
    const float* dyn_b1    = (const float*)d_in[12];
    const float* dyn_w2    = (const float*)d_in[13];
    const float* dyn_b2    = (const float*)d_in[14];
    const float* nca0_w1   = (const float*)d_in[15];
    const float* nca0_b1   = (const float*)d_in[16];
    const float* nca0_w2   = (const float*)d_in[17];
    const float* nca0_b2   = (const float*)d_in[18];
    const float* nca1_w1   = (const float*)d_in[19];
    const float* nca1_b1   = (const float*)d_in[20];
    const float* nca1_w2   = (const float*)d_in[21];
    const float* nca1_b2   = (const float*)d_in[22];
    const float* tcn_w1    = (const float*)d_in[23];
    const float* tcn_b1    = (const float*)d_in[24];
    const float* tcn_w2    = (const float*)d_in[25];
    const float* tcn_b2    = (const float*)d_in[26];
    const float* tcn_pw    = (const float*)d_in[27];
    const float* tcn_pb    = (const float*)d_in[28];
    const float* pred_w    = (const float*)d_in[29];
    const float* pred_b    = (const float*)d_in[30];
    const float* cls_w     = (const float*)d_in[31];
    const float* cls_b     = (const float*)d_in[32];
    float* out = (float*)d_out;
    char*  ws  = (char*)d_ws;

    const size_t BT = (size_t)BATCH * TT;   // 49152

    // ---- workspace layout (persistent, then aliased arena) ----
    size_t off = 0;
    auto alloc = [&](size_t bytes) { size_t r = off; off += alup(bytes); return r; };
    size_t oWdyn1 = alloc(256 * 256 * 2);
    size_t oWdyn2 = alloc(256 * 256 * 2);
    size_t oWn01  = alloc(160  * 256 * 2);
    size_t oWn02  = alloc(1280 * 256 * 2);
    size_t oWn11  = alloc(1280 * 256 * 2);
    size_t oWn12  = alloc(1280 * 256 * 2);
    size_t oWt1   = alloc(2048 * 256 * 2);
    size_t oWt2   = alloc(1024 * 256 * 2);
    size_t oWpj   = alloc(256  * 256 * 2);
    size_t oDyn   = alloc(BT * 256 * 2);
    size_t oXn    = alloc((size_t)BATCH * HIST * 256 * 2);
    size_t oStat  = alloc(BT * 256 * 2);
    size_t oWin   = alloc((size_t)BATCH * WROWS * 512 * 2);
    size_t oH1    = alloc((size_t)BATCH * TK * 256 * 2);
    size_t oH2    = alloc((size_t)BATCH * 256 * 2);
    size_t oPe    = alloc((size_t)BATCH * 256 * 2);
    size_t arena  = off;
    size_t oDynin = arena;
    size_t oHid   = arena + alup(BT * 256 * 2);
    size_t oY0    = arena;                                // after dyn MLP done
    size_t oY1    = arena + alup(BT * SS * 256 * 2);
    size_t oZ0    = arena;                                // after y0 dead

    bf16_t* Wdyn1 = (bf16_t*)(ws + oWdyn1);
    bf16_t* Wdyn2 = (bf16_t*)(ws + oWdyn2);
    bf16_t* Wn01  = (bf16_t*)(ws + oWn01);
    bf16_t* Wn02  = (bf16_t*)(ws + oWn02);
    bf16_t* Wn11  = (bf16_t*)(ws + oWn11);
    bf16_t* Wn12  = (bf16_t*)(ws + oWn12);
    bf16_t* Wt1   = (bf16_t*)(ws + oWt1);
    bf16_t* Wt2   = (bf16_t*)(ws + oWt2);
    bf16_t* Wpj   = (bf16_t*)(ws + oWpj);
    bf16_t* dyn   = (bf16_t*)(ws + oDyn);
    bf16_t* xnb   = (bf16_t*)(ws + oXn);
    bf16_t* statb = (bf16_t*)(ws + oStat);
    bf16_t* win   = (bf16_t*)(ws + oWin);
    bf16_t* h1    = (bf16_t*)(ws + oH1);
    bf16_t* h2    = (bf16_t*)(ws + oH2);
    bf16_t* pe    = (bf16_t*)(ws + oPe);
    bf16_t* dynin = (bf16_t*)(ws + oDynin);
    bf16_t* hid   = (bf16_t*)(ws + oHid);
    bf16_t* y0    = (bf16_t*)(ws + oY0);
    bf16_t* y1    = (bf16_t*)(ws + oY1);
    bf16_t* z0    = (bf16_t*)(ws + oZ0);

    dim3 blk(256);
    auto blocks = [](size_t n) { return dim3((unsigned)((n + 255) / 256)); };

    // ---- pack weights (fragment-major B layout) ----
    pack_dense_frag_kernel<<<blocks(256 * 256), blk, 0, stream>>>(Wdyn1, dyn_w1, 256, 256);
    pack_dense_frag_kernel<<<blocks(256 * 256), blk, 0, stream>>>(Wdyn2, dyn_w2, 256, 256);
    pack_conv_frag_kernel<<<blocks(160  * 256), blk, 0, stream>>>(Wn01, nca0_w1, INDIM, 5, 32, 256);
    pack_conv_frag_kernel<<<blocks(1280 * 256), blk, 0, stream>>>(Wn02, nca0_w2, 256, 5, 256, 256);
    pack_conv_frag_kernel<<<blocks(1280 * 256), blk, 0, stream>>>(Wn11, nca1_w1, 256, 5, 256, 256);
    pack_conv_frag_kernel<<<blocks(1280 * 256), blk, 0, stream>>>(Wn12, nca1_w2, 256, 5, 256, 256);
    pack_conv_frag_kernel<<<blocks(2048 * 256), blk, 0, stream>>>(Wt1, tcn_w1, 512, 4, 512, 256);
    pack_conv_frag_kernel<<<blocks(1024 * 256), blk, 0, stream>>>(Wt2, tcn_w2, 256, 4, 256, 256);
    pack_conv_frag_kernel<<<blocks(256  * 256), blk, 0, stream>>>(Wpj, tcn_pw, 256, 1, 256, 256);

    // ---- embeddings / tiny projections ----
    embed_kernel<<<blocks(BT * 256), blk, 0, stream>>>(dynin, city_locs, date_emb,
                                                       emb_month, emb_wd, emb_hour, loc_w, loc_b);
    xn_kernel<<<blocks((size_t)BATCH * HIST * 256), blk, 0, stream>>>(xnb, pm25_hist, pm25_w, pm25_b);

    // ---- dyn MLP: two tanh GEMMs (49152 x 256 x 256) ----
    { GP g{dynin, nullptr, 256, 0};
      gemm_kernel<0, 1><<<dim3(384, 4), blk, 0, stream>>>(g, Wdyn1, dyn_b1, hid, 256, 256); }
    { GP g{hid, nullptr, 256, 0};
      gemm_kernel<0, 1><<<dim3(384, 4), blk, 0, stream>>>(g, Wdyn2, dyn_b2, dyn, 256, 256); }

    // ---- nca0 conv1: all s (M = B*T*6 = 294912), K = 5*32 ----
    { GP g{nullptr, features, 0, 0};
      gemm_kernel<1, 2><<<dim3(2304, 4), blk, 0, stream>>>(g, Wn01, nca0_b1, y0, 160, 256); }
    // ---- nca0 conv2: s in {1,3,5} (M = B*T*3 = 147456), K = 5*256 ----
    { GP g{y0, nullptr, 0, 0};
      gemm_kernel<2, 2><<<dim3(1152, 4), blk, 0, stream>>>(g, Wn02, nca0_b2, y1, 1280, 256); }
    // ---- + dyn broadcast ----
    add_dyn_kernel<<<blocks((size_t)BATCH * TT * 3 * 256), blk, 0, stream>>>(y1, dyn);
    // ---- nca1 conv1 (dil=2): s in {1,3,5}, K = 5*256 ----
    { GP g{y1, nullptr, 0, 0};
      gemm_kernel<3, 2><<<dim3(1152, 4), blk, 0, stream>>>(g, Wn11, nca1_b1, z0, 1280, 256); }
    // ---- nca1 conv2 (dil=2): s=5 only -> static (M = B*T = 49152) ----
    { GP g{z0, nullptr, 0, 0};
      gemm_kernel<4, 2><<<dim3(384, 4), blk, 0, stream>>>(g, Wn12, nca1_b2, statb, 1280, 256); }

    // ---- decoder ----
    init_window_kernel<<<blocks((size_t)BATCH * RF * 512), blk, 0, stream>>>(win, statb, xnb);
    for (int step = 0; step < PRED; ++step) {
        { GP g{win, nullptr, 0, step};   // conv1: M = B*4 = 2048, K = 4*512
          gemm_kernel<5, 2><<<dim3(16, 4), blk, 0, stream>>>(g, Wt1, tcn_b1, h1, 2048, 256); }
        { GP g{h1, nullptr, 0, 0};       // conv2: M = B = 512, K = 4*256
          gemm_kernel<6, 2><<<dim3(4, 4), blk, 0, stream>>>(g, Wt2, tcn_b2, h2, 1024, 256); }
        { GP g{h2, nullptr, 256, 0};     // proj:  M = B, K = 256, no act
          gemm_kernel<0, 0><<<dim3(4, 4), blk, 0, stream>>>(g, Wpj, tcn_pb, pe, 256, 256); }
        finish_step_kernel<<<dim3(BATCH), blk, 0, stream>>>(pe, pred_w, pred_b, cls_w, cls_b,
                                                            pm25_w, pm25_b, statb, win, out, step);
    }
}
